// SceneEncoder_32212254720754
// MI455X (gfx1250) — matrix operand order repaired and checked by reading, not compiled, not run on hardware
//
#include <hip/hip_runtime.h>
#include <hip/hip_bf16.h>

#define A_N   128
#define T_N   50
#define L_N   256
#define NN    6400      // A_N * T_N
#define HID   128
#define FFN_D 512
#define NHEAD 8

typedef __attribute__((ext_vector_type(16))) _Float16 v16h;
typedef __attribute__((ext_vector_type(8)))  _Float16 v8h;
typedef __attribute__((ext_vector_type(8)))  float    v8f;
typedef __attribute__((ext_vector_type(4)))  float    v4f;

// ---------------- device helpers ----------------

__device__ inline float wave_sum32(float v) {
#pragma unroll
  for (int o = 16; o > 0; o >>= 1) v += __shfl_xor(v, o, 32);
  return v;
}

__device__ inline float wrap_pi(float a) {
  const float PI = 3.14159265358979323846f;
  float r = fmodf(a + PI, 2.0f * PI);
  if (r < 0.f) r += 2.0f * PI;
  return r - PI;
}

__device__ inline void atomicMaxF(float* addr, float val) {
  unsigned int* u = (unsigned int*)addr;
  unsigned int cur = __float_as_uint(*addr);
  while (__uint_as_float(cur) < val) {
    unsigned int old = atomicCAS(u, cur, __float_as_uint(val));
    if (old == cur) break;
    cur = old;
  }
}

// ---------------- WMMA GEMM ----------------
// C[M,N] = act(A[M,K] @ Wt^T + bias [+ residual])
// Wt: [N,K] f16 (pre-transposed weights). A element type via template (f32 or f16).
// Each wave computes a 16x64 strip: one A fragment feeds 4 WMMAs per k-step.
// act: 0=none, 1=relu, 2=sigmoid.  Output: f32 (Cf) or f16 (Ch).
template <bool AHALF>
__global__ void gemm_wmma_k(const void* __restrict__ Asrc,
                            const _Float16* __restrict__ Wt, const float* __restrict__ bias,
                            const float* __restrict__ residual,
                            float* __restrict__ Cf, _Float16* __restrict__ Ch,
                            int M, int N, int K, int act)
{
  const int wave = threadIdx.x >> 5;
  const int lane = threadIdx.x & 31;
  const int l15  = lane & 15;
  const int hsel = lane >> 4;                 // 0 or 1
  const int r0   = (blockIdx.x * 4 + wave) * 16;
  const int c0   = blockIdx.y * 64;
  const int row  = r0 + l15;
  const int rowc = (row < M) ? row : (M - 1); // clamp: keep EXEC all-1s for WMMA

  v8f acc0 = {0.f,0.f,0.f,0.f,0.f,0.f,0.f,0.f};
  v8f acc1 = acc0, acc2 = acc0, acc3 = acc0;

  const _Float16* bp0 = Wt + (size_t)(c0 + l15) * K;
  const _Float16* bp1 = bp0 + (size_t)16 * K;
  const _Float16* bp2 = bp0 + (size_t)32 * K;
  const _Float16* bp3 = bp0 + (size_t)48 * K;

  const float*    apf = AHALF ? nullptr : (const float*)Asrc + (size_t)rowc * K;
  const _Float16* aph = AHALF ? (const _Float16*)Asrc + (size_t)rowc * K : nullptr;

  for (int k0 = 0; k0 < K; k0 += 32) {
    const int ka = k0 + hsel * 16;
    v16h a;
    if (AHALF) {
      *((v8h*)&a)     = *(const v8h*)(aph + ka);
      *((v8h*)&a + 1) = *(const v8h*)(aph + ka + 8);
    } else {
      v4f a0 = *(const v4f*)(apf + ka);
      v4f a1 = *(const v4f*)(apf + ka + 4);
      v4f a2 = *(const v4f*)(apf + ka + 8);
      v4f a3 = *(const v4f*)(apf + ka + 12);
#pragma unroll
      for (int i = 0; i < 4; ++i) {
        a[i]      = (_Float16)a0[i];
        a[4 + i]  = (_Float16)a1[i];
        a[8 + i]  = (_Float16)a2[i];
        a[12 + i] = (_Float16)a3[i];
      }
    }
    const int kb = k0 + hsel * 16;
    v16h b0 = *(const v16h*)(bp0 + kb);
    v16h b1 = *(const v16h*)(bp1 + kb);
    v16h b2 = *(const v16h*)(bp2 + kb);
    v16h b3 = *(const v16h*)(bp3 + kb);
    acc0 = __builtin_amdgcn_wmma_f32_16x16x32_f16(false, a, false, b0, (short)0, acc0, false, false);
    acc1 = __builtin_amdgcn_wmma_f32_16x16x32_f16(false, a, false, b1, (short)0, acc1, false, false);
    acc2 = __builtin_amdgcn_wmma_f32_16x16x32_f16(false, a, false, b2, (short)0, acc2, false, false);
    acc3 = __builtin_amdgcn_wmma_f32_16x16x32_f16(false, a, false, b3, (short)0, acc3, false, false);
  }

  v8f accs[4] = {acc0, acc1, acc2, acc3};
#pragma unroll
  for (int t = 0; t < 4; ++t) {
    const int col = c0 + 16 * t + l15;
    const float bb = bias ? bias[col] : 0.f;
#pragma unroll
    for (int j = 0; j < 8; ++j) {
      const int m = r0 + hsel * 8 + j;
      if (m < M) {
        float r = accs[t][j] + bb;
        if (residual) r += residual[(size_t)m * N + col];
        if (act == 1) r = fmaxf(r, 0.f);
        else if (act == 2) r = 1.f / (1.f + __expf(-r));
        if (Cf) Cf[(size_t)m * N + col] = r;
        else    Ch[(size_t)m * N + col] = (_Float16)r;
      }
    }
  }
}

// Wt[n*K+k] = (f16) W[k*N+n]
__global__ void prep_wt(const float* __restrict__ W, _Float16* __restrict__ Wt, int K, int N) {
  int i = blockIdx.x * blockDim.x + threadIdx.x;
  if (i < K * N) {
    int k = i / N, n = i - k * N;
    Wt[(size_t)n * K + k] = (_Float16)W[i];
  }
}

// ---------------- LayerNorm (wave per row, HID=128) ----------------
__global__ void ln_kernel(const float* __restrict__ x, const float* __restrict__ g,
                          const float* __restrict__ b, float* __restrict__ y, int M)
{
  int row  = blockIdx.x * (blockDim.x >> 5) + (threadIdx.x >> 5);
  int lane = threadIdx.x & 31;
  if (row >= M) return;
  const float* xr = x + (size_t)row * HID;
  v4f v = *(const v4f*)(xr + lane * 4);
  float mu = wave_sum32(v.x + v.y + v.z + v.w) * (1.f / HID);
  float d0 = v.x - mu, d1 = v.y - mu, d2 = v.z - mu, d3 = v.w - mu;
  float var = wave_sum32(d0 * d0 + d1 * d1 + d2 * d2 + d3 * d3) * (1.f / HID);
  float rs = rsqrtf(var + 1e-5f);
  float* yr = y + (size_t)row * HID;
  int j = lane * 4;
  yr[j]     = d0 * rs * g[j]     + b[j];
  yr[j + 1] = d1 * rs * g[j + 1] + b[j + 1];
  yr[j + 2] = d2 * rs * g[j + 2] + b[j + 2];
  yr[j + 3] = d3 * rs * g[j + 3] + b[j + 3];
}

// ---------------- MLP first layer (tiny K) + LN + ReLU, f16 out ----------------
__global__ void mlp_head(const float* __restrict__ feat, int Kin,
                         const float* __restrict__ W1, const float* __restrict__ b1,
                         const float* __restrict__ ng, const float* __restrict__ nb,
                         _Float16* __restrict__ out, int M)
{
  int row  = blockIdx.x * (blockDim.x >> 5) + (threadIdx.x >> 5);
  int lane = threadIdx.x & 31;
  if (row >= M) return;
  float f[4];
  for (int c = 0; c < 4; ++c) f[c] = (c < Kin) ? feat[(size_t)row * Kin + c] : 0.f;
  float h[4];
#pragma unroll
  for (int i = 0; i < 4; ++i) {
    int j = lane * 4 + i;
    float acc = b1[j];
    for (int c = 0; c < Kin; ++c) acc += f[c] * W1[c * HID + j];
    h[i] = acc;
  }
  float mu = wave_sum32(h[0] + h[1] + h[2] + h[3]) * (1.f / HID);
  float sq = 0.f;
#pragma unroll
  for (int i = 0; i < 4; ++i) { float d = h[i] - mu; sq += d * d; }
  float var = wave_sum32(sq) * (1.f / HID);
  float rs = rsqrtf(var + 1e-5f);
  _Float16* o = out + (size_t)row * HID;
#pragma unroll
  for (int i = 0; i < 4; ++i) {
    int j = lane * 4 + i;
    float y = (h[i] - mu) * rs * ng[j] + nb[j];
    o[j] = (_Float16)fmaxf(y, 0.f);
  }
}

// ---------------- edge geometry features ----------------
// mode 0: t2a (4 feats, flat ids a*T+t) | mode 1: l2a (3 feats, src=lane)
// mode 2: a2a (3 feats, time-major ids t*A+a)
__global__ void edge_geo(const float* __restrict__ apos, const float* __restrict__ ahead,
                         const float* __restrict__ lpos, const float* __restrict__ lhead,
                         const int* __restrict__ eidx, int E, int mode,
                         float* __restrict__ feat)
{
  int e = blockIdx.x * blockDim.x + threadIdx.x;
  if (e >= E) return;
  int s = eidx[e], d = eidx[E + e];
  float sx, sy, sh, px, py, ph;
  if (mode == 0) {
    sx = apos[2 * s]; sy = apos[2 * s + 1]; sh = ahead[s];
    px = apos[2 * d]; py = apos[2 * d + 1]; ph = ahead[d];
  } else if (mode == 1) {
    sx = lpos[2 * s]; sy = lpos[2 * s + 1]; sh = lhead[s];
    px = apos[2 * d]; py = apos[2 * d + 1]; ph = ahead[d];
  } else {
    int ts = s / A_N, as = s - ts * A_N; int fs = as * T_N + ts;
    int td = d / A_N, ad = d - td * A_N; int fd = ad * T_N + td;
    sx = apos[2 * fs]; sy = apos[2 * fs + 1]; sh = ahead[fs];
    px = apos[2 * fd]; py = apos[2 * fd + 1]; ph = ahead[fd];
  }
  float ddx = sx - px, ddy = sy - py;
  float c = cosf(ph), sn = sinf(ph);
  float x = c * ddx + sn * ddy;
  float y = -sn * ddx + c * ddy;
  float el = sqrtf(x * x + y * y);
  float et = atan2f(y, x);
  float eh = wrap_pi(sh - ph);
  if (mode == 0) {
    feat[(size_t)e * 4]     = el; feat[(size_t)e * 4 + 1] = et;
    feat[(size_t)e * 4 + 2] = eh; feat[(size_t)e * 4 + 3] = (float)(s - d);
  } else {
    feat[(size_t)e * 3]     = el; feat[(size_t)e * 3 + 1] = et;
    feat[(size_t)e * 3 + 2] = eh;
  }
}

// ---------------- attention softmax pieces ----------------
__global__ void sim_kernel(const float* __restrict__ q, const float* __restrict__ k,
                           const _Float16* __restrict__ ke, const int* __restrict__ eidx,
                           int E, float* __restrict__ sim, float* __restrict__ mx)
{
  int t = blockIdx.x * blockDim.x + threadIdx.x;
  if (t >= E * NHEAD) return;
  int e = t >> 3, h = t & 7;
  int s = eidx[e], d = eidx[E + e];
  const float*    qp  = q  + (size_t)d * HID + h * 16;
  const float*    kp  = k  + (size_t)s * HID + h * 16;
  const _Float16* kep = ke + (size_t)e * HID + h * 16;
  float acc = 0.f;
#pragma unroll
  for (int i = 0; i < 16; ++i) acc += qp[i] * (kp[i] + (float)kep[i]);
  acc *= 0.25f;                       // 1/sqrt(16)
  sim[t] = acc;
  atomicMaxF(&mx[d * NHEAD + h], acc);
}

__global__ void expz_kernel(const float* __restrict__ sim, const float* __restrict__ mx,
                            const int* __restrict__ eidx, int E,
                            float* __restrict__ ew, float* __restrict__ z)
{
  int t = blockIdx.x * blockDim.x + threadIdx.x;
  if (t >= E * NHEAD) return;
  int e = t >> 3, h = t & 7;
  int d = eidx[E + e];
  float w = __expf(sim[t] - mx[d * NHEAD + h]);
  ew[t] = w;
  atomicAdd(&z[d * NHEAD + h], w);
}

__global__ void msg_kernel(const float* __restrict__ ew, const float* __restrict__ z,
                           const float* __restrict__ v, const _Float16* __restrict__ ve,
                           const int* __restrict__ eidx, int E, float* __restrict__ agg)
{
  int t = blockIdx.x * blockDim.x + threadIdx.x;
  if (t >= E * HID) return;
  int e = t >> 7, dd = t & 127, h = dd >> 4;
  int s = eidx[e], d = eidx[E + e];
  float attn = ew[(size_t)e * NHEAD + h] / z[d * NHEAD + h];
  float val = attn * (v[(size_t)s * HID + dd] + (float)ve[(size_t)e * HID + dd]);
  atomicAdd(&agg[(size_t)d * HID + dd], val);
}

// ---------------- elementwise ----------------
__global__ void fill_kernel(float* p, float v, int n) {
  int i = blockIdx.x * blockDim.x + threadIdx.x;
  if (i < n) p[i] = v;
}
__global__ void add_kernel(const float* a, const float* b, float* o, int n) {
  int i = blockIdx.x * blockDim.x + threadIdx.x;
  if (i < n) o[i] = a[i] + b[i];
}
__global__ void concat_kernel(const float* __restrict__ agg, const float* __restrict__ xd,
                              float* __restrict__ cat) {
  int i = blockIdx.x * blockDim.x + threadIdx.x;
  if (i >= NN * 256) return;
  int m = i >> 8, j = i & 255;
  cat[i] = (j < HID) ? agg[(size_t)m * HID + j] : xd[(size_t)m * HID + (j - HID)];
}
__global__ void gatecomb_kernel(const float* agg, const float* g, const float* s,
                                float* u, int n) {
  int i = blockIdx.x * blockDim.x + threadIdx.x;
  if (i < n) u[i] = agg[i] + g[i] * (s[i] - agg[i]);
}
__global__ void perm_kernel(const float* __restrict__ x, float* __restrict__ y, int to_tm) {
  int i = blockIdx.x * blockDim.x + threadIdx.x;
  if (i >= NN * HID) return;
  int node = i >> 7, ch = i & 127;
  int src;
  if (to_tm) { int t = node / A_N, a = node - t * A_N; src = a * T_N + t; }
  else       { int a = node / T_N, t = node - a * T_N; src = t * A_N + a; }
  y[i] = x[(size_t)src * HID + ch];
}

// ---------------- host side ----------------

struct AttnP {
  const float *W1, *W2, *Wg, *Wk, *Wke, *Wo, *Wq, *Ws, *Wv, *Wve;
  const float *b1, *b2, *bg, *bo, *bq, *bs, *n2b, *n2g, *ndb, *ndg, *nsb, *nsg;
};
struct MlpP { const float *W1, *W2, *b1, *b2, *nb, *ng; };

static AttnP attnAt(void* const* in, int base, bool bip) {
  AttnP p;
  p.W1  = (const float*)in[base + 0];  p.W2  = (const float*)in[base + 1];
  p.Wg  = (const float*)in[base + 2];  p.Wk  = (const float*)in[base + 3];
  p.Wke = (const float*)in[base + 4];  p.Wo  = (const float*)in[base + 5];
  p.Wq  = (const float*)in[base + 6];  p.Ws  = (const float*)in[base + 7];
  p.Wv  = (const float*)in[base + 8];  p.Wve = (const float*)in[base + 9];
  p.b1  = (const float*)in[base + 10]; p.b2  = (const float*)in[base + 11];
  p.bg  = (const float*)in[base + 12]; p.bo  = (const float*)in[base + 13];
  p.bq  = (const float*)in[base + 14]; p.bs  = (const float*)in[base + 15];
  p.n2b = (const float*)in[base + 16]; p.n2g = (const float*)in[base + 17];
  p.ndb = (const float*)in[base + 18]; p.ndg = (const float*)in[base + 19];
  p.nsb = bip ? (const float*)in[base + 20] : nullptr;
  p.nsg = bip ? (const float*)in[base + 21] : nullptr;
  return p;
}
static MlpP mlpAt(void* const* in, int base) {
  MlpP p;
  p.W1 = (const float*)in[base + 0]; p.W2 = (const float*)in[base + 1];
  p.b1 = (const float*)in[base + 2]; p.b2 = (const float*)in[base + 3];
  p.nb = (const float*)in[base + 4]; p.ng = (const float*)in[base + 5];
  return p;
}

// weight arena offsets in halves
enum { OQ = 0, OKK = 16384, OV = 32768, OKE = 49152, OVE = 65536,
       OS = 81920, OO = 98304, OG = 114688, O1 = 147456, O2 = 212992,
       WT_TOT = 278528 };

static inline int imax2(int a, int b) { return a > b ? a : b; }

static void prep(hipStream_t st, const float* W, _Float16* Wt, int K, int N) {
  int n = K * N;
  prep_wt<<<(n + 255) / 256, 256, 0, st>>>(W, Wt, K, N);
}
static void gemm(hipStream_t st, const float* Af, const _Float16* Ah, const _Float16* Wt,
                 const float* bias, const float* resid, float* Cf, _Float16* Ch,
                 int M, int N, int K, int act) {
  if (M <= 0) return;
  dim3 g((M + 63) / 64, N / 64);
  if (Ah)
    gemm_wmma_k<true><<<g, 128, 0, st>>>((const void*)Ah, Wt, bias, resid, Cf, Ch, M, N, K, act);
  else
    gemm_wmma_k<false><<<g, 128, 0, st>>>((const void*)Af, Wt, bias, resid, Cf, Ch, M, N, K, act);
}
static void ln_launch(hipStream_t st, const float* x, const float* g, const float* b,
                      float* y, int M) {
  ln_kernel<<<(M + 7) / 8, 256, 0, st>>>(x, g, b, y, M);
}

struct Ctx {
  hipStream_t st;
  float *xd, *xs, *q, *k, *v, *sb, *gb, *ub, *xb, *xn, *agg, *cat, *ffn, *mx, *z;
  float *simb, *ewb;
  _Float16 *eB, *eC, *wt;
};

static void run_attn(Ctx& c, const AttnP& p, const float* xsrc, int Ns,
                     const float* xdst, const int* eidx, int E,
                     const _Float16* attr, float* out)
{
  hipStream_t st = c.st;
  prep(st, p.Wq,  c.wt + OQ,  HID, HID);
  prep(st, p.Wk,  c.wt + OKK, HID, HID);
  prep(st, p.Wv,  c.wt + OV,  HID, HID);
  prep(st, p.Wke, c.wt + OKE, HID, HID);
  prep(st, p.Wve, c.wt + OVE, HID, HID);
  prep(st, p.Ws,  c.wt + OS,  HID, HID);
  prep(st, p.Wo,  c.wt + OO,  HID, HID);
  prep(st, p.Wg,  c.wt + OG,  256, HID);
  prep(st, p.W1,  c.wt + O1,  HID, FFN_D);
  prep(st, p.W2,  c.wt + O2,  FFN_D, HID);

  ln_launch(st, xdst, p.ndg, p.ndb, c.xd, NN);
  const float* xsp = c.xd;
  if (p.nsg) { ln_launch(st, xsrc, p.nsg, p.nsb, c.xs, Ns); xsp = c.xs; }

  gemm(st, c.xd, nullptr, c.wt + OQ,  p.bq,   nullptr, c.q,  nullptr, NN, HID, HID, 0);
  gemm(st, xsp,  nullptr, c.wt + OKK, nullptr, nullptr, c.k,  nullptr, Ns, HID, HID, 0);
  gemm(st, xsp,  nullptr, c.wt + OV,  nullptr, nullptr, c.v,  nullptr, Ns, HID, HID, 0);
  gemm(st, nullptr, attr, c.wt + OKE, nullptr, nullptr, nullptr, c.eB, E, HID, HID, 0);
  gemm(st, nullptr, attr, c.wt + OVE, nullptr, nullptr, nullptr, c.eC, E, HID, HID, 0);

  fill_kernel<<<(NN * NHEAD + 255) / 256, 256, 0, st>>>(c.mx, -1e30f, NN * NHEAD);
  fill_kernel<<<(NN * NHEAD + 255) / 256, 256, 0, st>>>(c.z, 0.f, NN * NHEAD);
  fill_kernel<<<(NN * HID + 255) / 256, 256, 0, st>>>(c.agg, 0.f, NN * HID);

  int n1 = E * NHEAD;
  sim_kernel <<<(n1 + 255) / 256, 256, 0, st>>>(c.q, c.k, c.eB, eidx, E, c.simb, c.mx);
  expz_kernel<<<(n1 + 255) / 256, 256, 0, st>>>(c.simb, c.mx, eidx, E, c.ewb, c.z);
  int n2 = E * HID;
  msg_kernel <<<(n2 + 255) / 256, 256, 0, st>>>(c.ewb, c.z, c.v, c.eC, eidx, E, c.agg);

  concat_kernel<<<(NN * 256 + 255) / 256, 256, 0, st>>>(c.agg, c.xd, c.cat);
  gemm(st, c.cat, nullptr, c.wt + OG, p.bg, nullptr, c.gb, nullptr, NN, HID, 256, 2);
  gemm(st, c.xd,  nullptr, c.wt + OS, p.bs, nullptr, c.sb, nullptr, NN, HID, HID, 0);
  gatecomb_kernel<<<(NN * HID + 255) / 256, 256, 0, st>>>(c.agg, c.gb, c.sb, c.ub, NN * HID);
  gemm(st, c.ub, nullptr, c.wt + OO, p.bo, xdst, c.xb, nullptr, NN, HID, HID, 0);
  ln_launch(st, c.xb, p.n2g, p.n2b, c.xn, NN);
  gemm(st, c.xn,  nullptr, c.wt + O1, p.b1, nullptr, c.ffn, nullptr, NN, FFN_D, HID, 1);
  gemm(st, c.ffn, nullptr, c.wt + O2, p.b2, c.xb,    out,   nullptr, NN, HID, FFN_D, 0);
}

static void run_edge_mlp(Ctx& c, const MlpP& m, const float* feat, int Kin, int E,
                         _Float16* attrOut, _Float16* wtE)
{
  mlp_head<<<(E + 7) / 8, 256, 0, c.st>>>(feat, Kin, m.W1, m.b1, m.ng, m.nb, c.eB, E);
  prep(c.st, m.W2, wtE, HID, HID);
  gemm(c.st, nullptr, c.eB, wtE, m.b2, nullptr, nullptr, attrOut, E, HID, HID, 0);
}

extern "C" void kernel_launch(void* const* d_in, const int* in_sizes, int n_in,
                              void* d_out, int out_size, void* d_ws, size_t ws_size,
                              hipStream_t stream)
{
  hipStream_t st = stream;
  const float* a_len  = (const float*)d_in[0];
  const float* a_pos  = (const float*)d_in[1];
  const float* a_head = (const float*)d_in[2];
  const float* l_pos  = (const float*)d_in[3];
  const float* l_head = (const float*)d_in[4];
  const float* l_emb  = (const float*)d_in[5];
  const int* e_t2a = (const int*)d_in[6];
  const int* e_l2a = (const int*)d_in[7];
  const int* e_a2a = (const int*)d_in[8];
  int E1 = in_sizes[6] / 2, E2 = in_sizes[7] / 2, E3 = in_sizes[8] / 2;
  int Emax = imax2(imax2(E1, E2), E3);
  int Erow = imax2(Emax, NN);

  // params flattened by sorted pytree keys
  AttnP a2a0 = attnAt(d_in, 9, false);
  AttnP a2a1 = attnAt(d_in, 29, false);
  MlpP  a2aE = mlpAt(d_in, 49);
  MlpP  aEmb = mlpAt(d_in, 55);
  AttnP l2aA = attnAt(d_in, 61, true);
  MlpP  l2aE = mlpAt(d_in, 83);
  AttnP t2aA = attnAt(d_in, 89, false);
  MlpP  t2aE = mlpAt(d_in, 109);

  // workspace carving (256B aligned)
  char* base = (char*)d_ws;
  size_t off = 0;
  auto carve = [&](size_t bytes) -> void* {
    void* r = base + off;
    off = (off + bytes + 255) & ~(size_t)255;
    return r;
  };
  const size_t NB = (size_t)NN * HID * sizeof(float);
  Ctx c; c.st = st;
  c.xd  = (float*)carve(NB);
  c.xs  = (float*)carve((size_t)L_N * HID * sizeof(float));
  c.q   = (float*)carve(NB);  c.k  = (float*)carve(NB);  c.v  = (float*)carve(NB);
  c.sb  = (float*)carve(NB);  c.gb = (float*)carve(NB);  c.ub = (float*)carve(NB);
  c.xb  = (float*)carve(NB);  c.xn = (float*)carve(NB);  c.agg = (float*)carve(NB);
  c.cat = (float*)carve((size_t)NN * 256 * sizeof(float));
  c.ffn = (float*)carve((size_t)NN * FFN_D * sizeof(float));
  c.mx  = (float*)carve((size_t)NN * NHEAD * sizeof(float));
  c.z   = (float*)carve((size_t)NN * NHEAD * sizeof(float));
  c.simb = (float*)carve((size_t)Emax * NHEAD * sizeof(float));
  c.ewb  = (float*)carve((size_t)Emax * NHEAD * sizeof(float));
  c.eB = (_Float16*)carve((size_t)Erow * HID * sizeof(_Float16));
  c.eC = (_Float16*)carve((size_t)Emax * HID * sizeof(_Float16));
  _Float16* eA      = (_Float16*)carve((size_t)Emax * HID * sizeof(_Float16));
  _Float16* attrA2A = (_Float16*)carve((size_t)E3 * HID * sizeof(_Float16));
  c.wt = (_Float16*)carve((size_t)WT_TOT * sizeof(_Float16));
  _Float16* wtE = (_Float16*)carve((size_t)HID * HID * sizeof(_Float16));
  float* featb  = (float*)carve((size_t)Emax * 4 * sizeof(float));
  float* a_embs = (float*)carve(NB);
  float* a_t    = (float*)carve(NB);
  float* a_l    = (float*)carve(NB);
  float* Xbuf   = (float*)carve(NB);
  float* Xt     = (float*)carve(NB);
  float* Xt2    = (float*)carve(NB);

  // --- agent embeddings: MLP(a_length) ---
  mlp_head<<<(NN + 7) / 8, 256, 0, st>>>(a_len, 1, aEmb.W1, aEmb.b1, aEmb.ng, aEmb.nb,
                                         c.eB, NN);
  prep(st, aEmb.W2, wtE, HID, HID);
  gemm(st, nullptr, c.eB, wtE, aEmb.b2, nullptr, a_embs, nullptr, NN, HID, HID, 0);

  // --- t2a block ---
  edge_geo<<<(E1 + 255) / 256, 256, 0, st>>>(a_pos, a_head, l_pos, l_head, e_t2a, E1, 0, featb);
  run_edge_mlp(c, t2aE, featb, 4, E1, eA, wtE);
  run_attn(c, t2aA, a_embs, NN, a_embs, e_t2a, E1, eA, a_t);

  // --- l2a block (bipartite: src = lanes) ---
  edge_geo<<<(E2 + 255) / 256, 256, 0, st>>>(a_pos, a_head, l_pos, l_head, e_l2a, E2, 1, featb);
  run_edge_mlp(c, l2aE, featb, 3, E2, eA, wtE);
  run_attn(c, l2aA, l_emb, L_N, a_embs, e_l2a, E2, eA, a_l);

  add_kernel<<<(NN * HID + 255) / 256, 256, 0, st>>>(a_t, a_l, Xbuf, NN * HID);
  perm_kernel<<<(NN * HID + 255) / 256, 256, 0, st>>>(Xbuf, Xt, 1);   // to time-major

  // --- a2a blocks (time-major node ids) ---
  edge_geo<<<(E3 + 255) / 256, 256, 0, st>>>(a_pos, a_head, l_pos, l_head, e_a2a, E3, 2, featb);
  run_edge_mlp(c, a2aE, featb, 3, E3, attrA2A, wtE);
  run_attn(c, a2a0, Xt, NN, Xt, e_a2a, E3, attrA2A, Xt2);
  run_attn(c, a2a1, Xt2, NN, Xt2, e_a2a, E3, attrA2A, Xt);

  perm_kernel<<<(NN * HID + 255) / 256, 256, 0, st>>>(Xt, (float*)d_out, 0); // back to agent-major
  hipMemcpyAsync((float*)d_out + (size_t)NN * HID, l_emb,
                 (size_t)L_N * HID * sizeof(float), hipMemcpyDeviceToDevice, st);

  (void)n_in; (void)out_size; (void)ws_size;
}
